// DotProductAttention_16312285790468
// MI455X (gfx1250) — compile-verified, matrix operand order checked by reading, not run on hardware
//
#include <hip/hip_runtime.h>

typedef __attribute__((ext_vector_type(16))) _Float16 v16h;
typedef __attribute__((ext_vector_type(4)))  _Float16 v4h;
typedef __attribute__((ext_vector_type(8)))  float    v8f;

// Problem constants (match reference)
constexpr int Bb = 8;
constexpr int Qq = 2048;
constexpr int Kk = 2048;
constexpr int Dd = 64;

constexpr int WAVES = 4;                 // waves per block (wave32)
constexpr int BLOCK = WAVES * 32;        // 128 threads
constexpr int QT    = WAVES * 16;        // 64 query rows per block
constexpr int KT    = 64;                // key tile per iteration
constexpr int LDK   = 72;                // padded row stride (halfs), 144B = 16B-aligned rows
constexpr int LDV   = 72;                // padded row stride (halfs) for transposed V
constexpr int LDP   = 72;                // padded row stride (halfs) for P scratch

// DPP row_ror-based reductions over the 16-lane half-rows (wave32 rows of 16).
// Rotation-tree reduction: after ror 8,4,2,1 every lane holds the full
// 16-lane reduction. VALU-only -> co-executes with WMMA, no LDS-pipe traffic.
template <int CTRL>
__device__ __forceinline__ float dpp_mov(float x) {
    int r = __builtin_amdgcn_update_dpp(0, __builtin_bit_cast(int, x),
                                        CTRL, 0xF, 0xF, true);
    return __builtin_bit_cast(float, r);
}
__device__ __forceinline__ float rowmax16(float r) {
    r = fmaxf(r, dpp_mov<0x128>(r));   // row_ror:8
    r = fmaxf(r, dpp_mov<0x124>(r));   // row_ror:4
    r = fmaxf(r, dpp_mov<0x122>(r));   // row_ror:2
    r = fmaxf(r, dpp_mov<0x121>(r));   // row_ror:1
    return r;
}
__device__ __forceinline__ float rowsum16(float r) {
    r += dpp_mov<0x128>(r);
    r += dpp_mov<0x124>(r);
    r += dpp_mov<0x122>(r);
    r += dpp_mov<0x121>(r);
    return r;
}

__global__ __launch_bounds__(BLOCK)
void flash_attn_f16wmma(const float* __restrict__ qg,
                        const float* __restrict__ kg,
                        const float* __restrict__ vg,
                        const int*   __restrict__ vlen,
                        float*       __restrict__ outg)
{
    __shared__ _Float16 Ksh[KT][LDK];          // row-major: [key][d]
    __shared__ _Float16 Vsh[Dd][LDV];          // transposed: [d][key]
    __shared__ _Float16 Psh[WAVES][16][LDP];   // per-wave P scratch: [row][key]

    const int b    = blockIdx.y;
    const int wave = threadIdx.x >> 5;
    const int lane = threadIdx.x & 31;
    const int hg   = lane >> 4;      // half-group: 0 for lanes 0-15, 1 for 16-31
    const int ln   = lane & 15;
    const int qbase = blockIdx.x * QT + wave * 16;
    const int vl    = vlen[b];

    const float* qb = qg + (size_t)b * Qq * Dd;
    const float* kb = kg + (size_t)b * Kk * Dd;
    const float* vb = vg + (size_t)b * Kk * Dd;

    // 16-bit A/B fragment K-dim mapping (ISA 7.12.2):
    // element i (0..15) -> K = (i<8 ? i : i+8) + 8*hg
    auto kmap = [&](int i) { return (i < 8 ? i : i + 8) + 8 * hg; };

    // ---- Load Q A-fragments once (row = qbase + ln), fold in 1/sqrt(D) ----
    v16h a0, a1;
    {
        const float* qrow = qb + (size_t)(qbase + ln) * Dd;
        #pragma unroll
        for (int i = 0; i < 16; ++i) {
            const int d0 = kmap(i);
            a0[i] = (_Float16)(qrow[d0]      * 0.125f);
            a1[i] = (_Float16)(qrow[d0 + 32] * 0.125f);
        }
    }

    v8f o0 = {0,0,0,0,0,0,0,0};
    v8f o1 = {0,0,0,0,0,0,0,0};
    v8f o2 = {0,0,0,0,0,0,0,0};
    v8f o3 = {0,0,0,0,0,0,0,0};
    float mrow[8], lrow[8];
    #pragma unroll
    for (int j = 0; j < 8; ++j) { mrow[j] = -1e30f; lrow[j] = 0.0f; }

    for (int k0 = 0; k0 < Kk; k0 += KT) {
        // ---- Stage K (row-major) and V (transposed) tiles into LDS as f16 ----
        // KT*Dd/4 = 1024 float4 loads per matrix; 8 per thread, coalesced b128.
        #pragma unroll
        for (int t = threadIdx.x; t < KT * Dd / 4; t += BLOCK) {
            const int r  = t >> 4;          // key row (16 float4 per row)
            const int c4 = (t & 15) * 4;    // d column
            const float4 kf = *(const float4*)(kb + (size_t)(k0 + r) * Dd + c4);
            const float4 vf = *(const float4*)(vb + (size_t)(k0 + r) * Dd + c4);
            v4h kh;
            kh[0] = (_Float16)kf.x; kh[1] = (_Float16)kf.y;
            kh[2] = (_Float16)kf.z; kh[3] = (_Float16)kf.w;
            *(v4h*)&Ksh[r][c4] = kh;        // one b64 store
            Vsh[c4 + 0][r] = (_Float16)vf.x;
            Vsh[c4 + 1][r] = (_Float16)vf.y;
            Vsh[c4 + 2][r] = (_Float16)vf.z;
            Vsh[c4 + 3][r] = (_Float16)vf.w;
        }
        // Prefetch next tile into the cache hierarchy (global_prefetch_b8)
        if (k0 + KT < Kk) {
            __builtin_prefetch(kb + (size_t)(k0 + KT) * Dd + threadIdx.x * 32, 0, 0);
            __builtin_prefetch(vb + (size_t)(k0 + KT) * Dd + threadIdx.x * 32, 0, 0);
        }
        __syncthreads();

        // ---- S = Q @ K^T : four 16x16 column tiles, 8 WMMAs ----
        v8f s[4];
        #pragma unroll
        for (int n = 0; n < 4; ++n) {
            // B-fragment: column = key (k0 + 16n + ln); K-dim = d (contiguous)
            v16h bk0, bk1;
            const _Float16* krow = &Ksh[16 * n + ln][0];
            #pragma unroll
            for (int vv = 0; vv < 8; ++vv) {
                const int d0 = (vv < 4 ? 2 * vv : 16 + 2 * (vv - 4)) + 8 * hg;
                bk0[2 * vv]     = krow[d0];
                bk0[2 * vv + 1] = krow[d0 + 1];
                bk1[2 * vv]     = krow[d0 + 32];
                bk1[2 * vv + 1] = krow[d0 + 33];
            }
            v8f c = {0,0,0,0,0,0,0,0};
            c = __builtin_amdgcn_wmma_f32_16x16x32_f16(false, a0, false, bk0,
                                                       (short)0, c, false, false);
            c = __builtin_amdgcn_wmma_f32_16x16x32_f16(false, a1, false, bk1,
                                                       (short)0, c, false, false);
            s[n] = c;
        }

        // ---- Mask (d2l-style: -1e6 past valid_len); C-layout: col = k0+16n+ln ----
        #pragma unroll
        for (int n = 0; n < 4; ++n) {
            const bool ok = (k0 + 16 * n + ln) < vl;
            #pragma unroll
            for (int j = 0; j < 8; ++j)
                s[n][j] = ok ? s[n][j] : -1.0e6f;
        }

        // ---- Online softmax: row stats per half-group (M = j + 8*hg) ----
        float alpha[8];
        #pragma unroll
        for (int j = 0; j < 8; ++j) {
            float r = fmaxf(fmaxf(s[0][j], s[1][j]), fmaxf(s[2][j], s[3][j]));
            r = rowmax16(r);
            const float mnew = fmaxf(mrow[j], r);
            alpha[j] = __expf(mrow[j] - mnew);
            mrow[j]  = mnew;
        }
        #pragma unroll
        for (int j = 0; j < 8; ++j) {
            float rsum = 0.0f;
            #pragma unroll
            for (int n = 0; n < 4; ++n) {
                const float p = __expf(s[n][j] - mrow[j]);
                s[n][j] = p;
                rsum += p;
            }
            rsum = rowsum16(rsum);
            lrow[j] = lrow[j] * alpha[j] + rsum;
        }
        #pragma unroll
        for (int j = 0; j < 8; ++j) {
            o0[j] *= alpha[j]; o1[j] *= alpha[j];
            o2[j] *= alpha[j]; o3[j] *= alpha[j];
        }

        // ---- C-layout -> A-layout for P via per-wave LDS scratch ----
        #pragma unroll
        for (int n = 0; n < 4; ++n)
            #pragma unroll
            for (int j = 0; j < 8; ++j)
                Psh[wave][j + 8 * hg][16 * n + ln] = (_Float16)s[n][j];
        // DS ops from one wave are in-order (DScnt); same-wave RAW is safe.

        v16h pa0, pa1;   // P A-fragments: keys 0..31 and 32..63 of this tile
        {
            const _Float16* prow = &Psh[wave][ln][0];
            #pragma unroll
            for (int vv = 0; vv < 8; ++vv) {
                const int c0 = (vv < 4 ? 2 * vv : 16 + 2 * (vv - 4)) + 8 * hg;
                pa0[2 * vv]     = prow[c0];
                pa0[2 * vv + 1] = prow[c0 + 1];
                pa1[2 * vv]     = prow[c0 + 32];
                pa1[2 * vv + 1] = prow[c0 + 33];
            }
        }

        // ---- O += P @ V : 4 column groups x 2 key chunks = 8 WMMAs ----
        // V is transposed in LDS, so the K-dim (keys) is contiguous:
        // each B-fragment is two 16B-aligned b128 LDS loads.
        #pragma unroll
        for (int g = 0; g < 4; ++g) {
            v16h bv0, bv1;
            const _Float16* vrow = &Vsh[16 * g + ln][0];
            #pragma unroll
            for (int i = 0; i < 16; ++i) {
                const int kk = kmap(i);
                bv0[i] = vrow[kk];
                bv1[i] = vrow[kk + 32];
            }
            v8f* og = (g == 0) ? &o0 : (g == 1) ? &o1 : (g == 2) ? &o2 : &o3;
            v8f acc = *og;
            acc = __builtin_amdgcn_wmma_f32_16x16x32_f16(false, pa0, false, bv0,
                                                         (short)0, acc, false, false);
            acc = __builtin_amdgcn_wmma_f32_16x16x32_f16(false, pa1, false, bv1,
                                                         (short)0, acc, false, false);
            *og = acc;
        }
        __syncthreads();
    }

    // ---- Epilogue: divide by l, store f32 (C-layout: M = j + 8*hg, N = ln) ----
    float* ob = outg + ((size_t)b * Qq + qbase) * Dd;
    #pragma unroll
    for (int j = 0; j < 8; ++j) {
        const int M = j + 8 * hg;
        const float inv = 1.0f / lrow[j];
        float* orow = ob + (size_t)M * Dd;
        orow[ 0 + ln] = o0[j] * inv;
        orow[16 + ln] = o1[j] * inv;
        orow[32 + ln] = o2[j] * inv;
        orow[48 + ln] = o3[j] * inv;
    }
}

extern "C" void kernel_launch(void* const* d_in, const int* in_sizes, int n_in,
                              void* d_out, int out_size, void* d_ws, size_t ws_size,
                              hipStream_t stream) {
    const float* q    = (const float*)d_in[0];
    const float* k    = (const float*)d_in[1];
    const float* v    = (const float*)d_in[2];
    const int*   vlen = (const int*)d_in[3];
    float*       out  = (float*)d_out;

    dim3 grid(Qq / QT, Bb, 1);
    dim3 block(BLOCK, 1, 1);
    flash_attn_f16wmma<<<grid, block, 0, stream>>>(q, k, v, vlen, out);
}
